// decoder_52682068853335
// MI455X (gfx1250) — matrix operand order repaired and checked by reading, not compiled, not run on hardware
//
#include <hip/hip_runtime.h>
#include <hip/hip_bf16.h>

// ---------------------------------------------------------------------------
// Problem constants (from reference): B=8, C=256, O=128, H=W=64
// ---------------------------------------------------------------------------
#define BB    8
#define CC    256
#define OO    128
#define HH    64
#define WWID  64
#define HWPX  4096          // H*W
#define KTOT  2304          // C*9, reordered as K' = k9*256 + c (tap-major)

#define AS1 __attribute__((address_space(1)))
#define AS3 __attribute__((address_space(3)))

#if __has_builtin(__builtin_amdgcn_global_load_async_to_lds_b32)
#define HAVE_ASYNC_LDS 1
#else
#define HAVE_ASYNC_LDS 0
#endif

typedef __attribute__((ext_vector_type(16))) __bf16 v16bf;
typedef __attribute__((ext_vector_type(8)))  __bf16 v8bf;
typedef __attribute__((ext_vector_type(8)))  float  v8f;

__device__ __forceinline__ __bf16 f2bf(float f) {           // RNE (prep kernel)
  unsigned u = __builtin_bit_cast(unsigned, f);
  u += 0x7fffu + ((u >> 16) & 1u);
  unsigned short s = (unsigned short)(u >> 16);
  return __builtin_bit_cast(__bf16, s);
}

// pack two floats to packed bf16x2 (round-to-nearest, single v_perm_b32)
__device__ __forceinline__ unsigned pack_bf16(float flo, float fhi) {
  unsigned u0 = __builtin_bit_cast(unsigned, flo) + 0x8000u;
  unsigned u1 = __builtin_bit_cast(unsigned, fhi) + 0x8000u;
  return __builtin_amdgcn_perm(u1, u0, 0x07060302u);  // {u1.hi16, u0.hi16}
}

__device__ __forceinline__ void wait_async() {
#if __has_builtin(__builtin_amdgcn_s_wait_asynccnt)
  __builtin_amdgcn_s_wait_asynccnt(0);
#else
  asm volatile("s_wait_asynccnt 0x0" ::: "memory");
#endif
}

// ---------------------------------------------------------------------------
// fp32 -> bf16 weight conversion with K-reorder:
//   dst[m][k9*256 + c] = src[m][c*9 + k9]
// ---------------------------------------------------------------------------
__global__ void cvt_permute_bf16(const float* __restrict__ src,
                                 __bf16* __restrict__ dst, int total) {
  int i = blockIdx.x * blockDim.x + threadIdx.x;
  if (i >= total) return;
  int m  = i / KTOT;
  int r  = i - m * KTOT;        // K' index
  int c  = r & 255;
  int k9 = r >> 8;
  dst[i] = f2bf(src[(size_t)m * KTOT + c * 9 + k9]);
}

// ---------------------------------------------------------------------------
// Adaptive Gaussian guide kernel: kern[b,k,p] = exp(-0.5 * sum_c dg^2)
// 16x16 pixel tile per block; per channel the 18x18 halo tile is brought in
// with GLOBAL_LOAD_ASYNC_TO_LDS_B32 (ASYNCcnt) — border cells are zeroed once
// and never rewritten (zero padding is position-, not channel-, dependent).
// ---------------------------------------------------------------------------
__global__ void guide_kernel(const float* __restrict__ guide,
                             float* __restrict__ kern) {
  __shared__ float tile[18 * 18];
  const int b   = blockIdx.z;
  const int tx  = blockIdx.x * 16, ty = blockIdx.y * 16;
  const int lx  = threadIdx.x & 15, ly = threadIdx.x >> 4;
  const int tid = threadIdx.x;

  // staging geometry (constant across channels): up to 2 slots per thread
  const int i0 = tid, i1 = tid + 256;
  const int r0 = i0 / 18, q0 = i0 - r0 * 18;
  const int gh0 = ty - 1 + r0, gw0 = tx - 1 + q0;
  const bool inb0 = (gh0 >= 0 && gh0 < HH && gw0 >= 0 && gw0 < WWID);
  const int goff0 = gh0 * WWID + gw0;
  const int r1 = i1 / 18, q1 = i1 - r1 * 18;
  const int gh1 = ty - 1 + r1, gw1 = tx - 1 + q1;
  const bool inb1 = (i1 < 18 * 18) &&
                    (gh1 >= 0 && gh1 < HH && gw1 >= 0 && gw1 < WWID);
  const int goff1 = gh1 * WWID + gw1;

  for (int i = tid; i < 18 * 18; i += 256) tile[i] = 0.f;   // zero border once
  __syncthreads();

  float acc[9];
#pragma unroll
  for (int k = 0; k < 9; ++k) acc[k] = 0.f;

  for (int c = 0; c < CC; ++c) {
    const float* g = guide + ((size_t)(b * CC + c)) * HWPX;
#if HAVE_ASYNC_LDS
    if (inb0)
      __builtin_amdgcn_global_load_async_to_lds_b32(
          (AS1 int*)(g + goff0), (AS3 int*)&tile[i0], 0, 0);
    if (inb1)
      __builtin_amdgcn_global_load_async_to_lds_b32(
          (AS1 int*)(g + goff1), (AS3 int*)&tile[i1], 0, 0);
    wait_async();
#else
    if (inb0) tile[i0] = g[goff0];
    if (inb1) tile[i1] = g[goff1];
#endif
    __syncthreads();
    float ctr = tile[(ly + 1) * 18 + (lx + 1)];
#pragma unroll
    for (int k = 0; k < 9; ++k) {
      float d = tile[(ly + k / 3) * 18 + (lx + k % 3)] - ctr;
      acc[k] += d * d;
    }
    __syncthreads();   // WAR: next channel's writes vs this channel's reads
  }
  const int p = (ty + ly) * WWID + tx + lx;
#pragma unroll
  for (int k = 0; k < 9; ++k)
    kern[((size_t)(b * 9 + k)) * HWPX + p] = __expf(-0.5f * acc[k]);
}

// ---------------------------------------------------------------------------
// Implicit-GEMM conv via V_WMMA_F32_16X16X32_BF16, tap-major K ordering,
// double-buffered LDS B tile (single barrier per K-step).
//   out[b,m,p] = sum_{k9,c} X[b,c,nbr_k9(p)] * (MOD ? kern[b,k9,p] : 1) * Wt[m,k9*256+c]
// Block = 256 threads (8 waves); M_block = 8*16*MT, N_block = 32 pixels.
// Staging: each thread owns one pixel column and K'-pairs {2s,2s+1},{2s+16,2s+17}
// so each bf16 pair is packed with one v_perm_b32 and stored with one
// ds_store_b32 directly in this kernel's contiguous-K B layout (lanes 0-15: K 0-15,
// lanes 16-31: K 16-31 -> one contiguous 32B LDS read per lane per subtile).
// ---------------------------------------------------------------------------
template <bool MOD, int MT>
__global__ void pac_gemm(const __bf16* __restrict__ Wt,    // [Mtotal, KTOT] (K' order)
                         const void*  __restrict__ Xv,     // MOD: f32 [B,C,HW] else bf16
                         const float* __restrict__ kern,   // [B,9,HW] (MOD only)
                         float* __restrict__ out,          // [B,Mtotal,HW]
                         int Mtotal) {
  __shared__ __align__(32) __bf16 smemB[2][2][32 * 16];    // [parity][nsub][...]

  const int tid     = threadIdx.x;
  const int lane    = tid & 31;
  const int wave    = tid >> 5;
  const int b       = blockIdx.z;
  const int pixBase = blockIdx.x * 32;
  const int oBase   = wave * 16 * MT;
  const int half    = lane >> 4;

  // staging coordinates: pixel column sn (0..31), K'-pair base s2 (0,2,..,14)
  const int sn   = tid & 31;
  const int s2   = (tid >> 5) * 2;
  const int sp   = pixBase + sn;
  const int sph  = sp >> 6;
  const int spw  = sp & (WWID - 1);
  const int nsub = sn >> 4;
  const int nl   = sn & 15;

  const float*  Xf = (const float*)Xv;
  const __bf16* Xb = (const __bf16*)Xv;

  v8f acc[MT][2];
#pragma unroll
  for (int mt = 0; mt < MT; ++mt) {
    acc[mt][0] = (v8f){};
    acc[mt][1] = (v8f){};
  }

  int par = 0;
  for (int k9 = 0; k9 < 9; ++k9) {
    // tap geometry: uniform across block -> scalar
    const int dy = k9 / 3 - 1, dx = k9 % 3 - 1;
    const int hh = sph + dy, wp = spw + dx;
    const bool inb = ((unsigned)hh < (unsigned)HH) & ((unsigned)wp < (unsigned)WWID);
    const int hc = min(max(hh, 0), HH - 1);
    const int wc = min(max(wp, 0), WWID - 1);
    float kv = 0.f;
    if constexpr (MOD) {
      float kraw = kern[((size_t)(b * 9 + k9)) * HWPX + sp];
      kv = inb ? kraw : 0.f;                 // folds zero padding into the gain
    }
    // base address for this thread's staging column at c = s2
    const size_t xoff = ((size_t)(b * CC + s2)) * HWPX + hc * WWID + wc;

    for (int cb = 0; cb < CC; cb += 32) {
      const int kbK = k9 * 256 + cb;         // K' block base (contiguous overall)

      // ---- stage 32(K') x 32(N): two packed-bf16 pairs per thread ----
      unsigned pk0, pk1;
      if constexpr (MOD) {
        float v0 = Xf[xoff + (size_t)(cb + 0)  * HWPX] * kv;
        float v1 = Xf[xoff + (size_t)(cb + 1)  * HWPX] * kv;
        float v2 = Xf[xoff + (size_t)(cb + 16) * HWPX] * kv;
        float v3 = Xf[xoff + (size_t)(cb + 17) * HWPX] * kv;
        pk0 = pack_bf16(v0, v1);
        pk1 = pack_bf16(v2, v3);
      } else {
        unsigned a0 = __builtin_bit_cast(unsigned short, Xb[xoff + (size_t)(cb + 0)  * HWPX]);
        unsigned a1 = __builtin_bit_cast(unsigned short, Xb[xoff + (size_t)(cb + 1)  * HWPX]);
        unsigned a2 = __builtin_bit_cast(unsigned short, Xb[xoff + (size_t)(cb + 16) * HWPX]);
        unsigned a3 = __builtin_bit_cast(unsigned short, Xb[xoff + (size_t)(cb + 17) * HWPX]);
        pk0 = inb ? (a0 | (a1 << 16)) : 0u;
        pk1 = inb ? (a2 | (a3 << 16)) : 0u;
      }
      // dense-B layout: pair {2s,2s+1} -> row nl, cols s2..s2+1 (b32 aligned);
      //                 pair {2s+16,..} -> row nl+16
      *(unsigned*)&smemB[par][nsub][nl * 16 + s2]        = pk0;
      *(unsigned*)&smemB[par][nsub][(nl + 16) * 16 + s2] = pk1;

      __syncthreads();   // stage(i) visible; buffer i-2 reads already fenced

      // ---- B fragments: one contiguous 32B LDS read per subtile ----
      v16bf bfrag0 = *(const v16bf*)(&smemB[par][0][lane * 16]);
      v16bf bfrag1 = *(const v16bf*)(&smemB[par][1][lane * 16]);

      // ---- A fragments + WMMAs ----
#pragma unroll
      for (int mt = 0; mt < MT; ++mt) {
        const int mRow = oBase + mt * 16 + (lane & 15);
        const __bf16* wrow = Wt + (size_t)mRow * KTOT + kbK;
        union { v16bf v; v8bf h[2]; } a;
        a.h[0] = *(const v8bf*)(wrow + 16 * half);      // elems 0..7 : K=16*half..
        a.h[1] = *(const v8bf*)(wrow + 16 * half + 8);  // elems 8..15: K=16*half+8..
        if (kbK + 32 < KTOT) __builtin_prefetch(wrow + 32, 0, 3);

        acc[mt][0] = __builtin_amdgcn_wmma_f32_16x16x32_bf16(
            false, a.v, false, bfrag0, (short)0, acc[mt][0], false, false);
        acc[mt][1] = __builtin_amdgcn_wmma_f32_16x16x32_bf16(
            false, a.v, false, bfrag1, (short)0, acc[mt][1], false, false);
      }
      par ^= 1;
    }
  }

  // C/D layout: VGPR r -> M = 8*half + r, N = lane & 15
#pragma unroll
  for (int mt = 0; mt < MT; ++mt) {
#pragma unroll
    for (int ns = 0; ns < 2; ++ns) {
#pragma unroll
      for (int r = 0; r < 8; ++r) {
        int m = oBase + mt * 16 + 8 * half + r;
        out[((size_t)(b * Mtotal + m)) * HWPX + pixBase + ns * 16 + (lane & 15)] =
            acc[mt][ns][r];
      }
    }
  }
}

// ---------------------------------------------------------------------------
// InstanceNorm (biased var, eps=1e-5) + ReLU; one block per (b, channel).
// float4 loads cached in registers across the stats phase (single read of x).
// ---------------------------------------------------------------------------
template <bool BF16OUT>
__global__ void instnorm_relu(const float* __restrict__ x, void* __restrict__ yv) {
  const size_t base = (size_t)blockIdx.x * HWPX;
  const float4* p4 = (const float4*)(x + base);

  float4 v4[4];
  float s = 0.f, ss = 0.f;
#pragma unroll
  for (int j = 0; j < 4; ++j) {
    float4 v = p4[threadIdx.x + j * 256];
    v4[j] = v;
    s  += (v.x + v.y) + (v.z + v.w);
    ss += (v.x * v.x + v.y * v.y) + (v.z * v.z + v.w * v.w);
  }
  __shared__ float rs[256], rss[256];
  rs[threadIdx.x] = s;
  rss[threadIdx.x] = ss;
  __syncthreads();
  for (int off = 128; off > 0; off >>= 1) {
    if (threadIdx.x < off) {
      rs[threadIdx.x]  += rs[threadIdx.x + off];
      rss[threadIdx.x] += rss[threadIdx.x + off];
    }
    __syncthreads();
  }
  const float mean = rs[0] * (1.f / HWPX);
  const float var  = rss[0] * (1.f / HWPX) - mean * mean;
  const float rstd = rsqrtf(var + 1e-5f);

#pragma unroll
  for (int j = 0; j < 4; ++j) {
    float4 v = v4[j];
    v.x = fmaxf((v.x - mean) * rstd, 0.f);
    v.y = fmaxf((v.y - mean) * rstd, 0.f);
    v.z = fmaxf((v.z - mean) * rstd, 0.f);
    v.w = fmaxf((v.w - mean) * rstd, 0.f);
    if constexpr (BF16OUT) {
      uint2 o;
      o.x = pack_bf16(v.x, v.y);
      o.y = pack_bf16(v.z, v.w);
      ((uint2*)((__bf16*)yv + base))[threadIdx.x + j * 256] = o;
    } else {
      ((float4*)((float*)yv + base))[threadIdx.x + j * 256] = v;
    }
  }
}

// ---------------------------------------------------------------------------
// Host launch
// ---------------------------------------------------------------------------
extern "C" void kernel_launch(void* const* d_in, const int* in_sizes, int n_in,
                              void* d_out, int out_size, void* d_ws, size_t ws_size,
                              hipStream_t stream) {
  const float* feat  = (const float*)d_in[0];   // [8,256,64,64]
  const float* guide = (const float*)d_in[1];   // [8,256,64,64]
  const float* pac_w = (const float*)d_in[2];   // [256,256,3,3]
  const float* dec_w = (const float*)d_in[3];   // [128,256,3,3]
  // d_in[4] (dec_b) cancels exactly under the following InstanceNorm.
  (void)in_sizes; (void)n_in; (void)out_size; (void)ws_size;

  char* w = (char*)d_ws;
  size_t off = 0;
  auto alloc = [&](size_t bytes) -> void* {
    off = (off + 255) & ~(size_t)255;
    void* p = w + off;
    off += bytes;
    return p;
  };
  float*  kern   = (float*) alloc((size_t)BB * 9 * HWPX * sizeof(float));
  __bf16* pacWb  = (__bf16*)alloc((size_t)CC * KTOT * sizeof(__bf16));
  __bf16* decWb  = (__bf16*)alloc((size_t)OO * KTOT * sizeof(__bf16));
  float*  pacOut = (float*) alloc((size_t)BB * CC * HWPX * sizeof(float));
  __bf16* feat2  = (__bf16*)alloc((size_t)BB * CC * HWPX * sizeof(__bf16));
  float*  yTmp   = (float*) alloc((size_t)BB * OO * HWPX * sizeof(float));

  const int nPac = CC * KTOT;   // 589824
  const int nDec = OO * KTOT;   // 294912
  cvt_permute_bf16<<<(nPac + 255) / 256, 256, 0, stream>>>(pac_w, pacWb, nPac);
  cvt_permute_bf16<<<(nDec + 255) / 256, 256, 0, stream>>>(dec_w, decWb, nDec);

  guide_kernel<<<dim3(WWID / 16, HH / 16, BB), 256, 0, stream>>>(guide, kern);

  // PAC conv: M=256 = 8 waves x 2 subtiles x 16 rows, N_block = 32 px
  pac_gemm<true, 2><<<dim3(HWPX / 32, 1, BB), 256, 0, stream>>>(
      pacWb, (const void*)feat, kern, pacOut, CC);

  instnorm_relu<true><<<BB * CC, 256, 0, stream>>>(pacOut, (void*)feat2);

  // Decoder conv: M=128 = 8 waves x 1 subtile x 16 rows
  pac_gemm<false, 1><<<dim3(HWPX / 32, 1, BB), 256, 0, stream>>>(
      decWb, (const void*)feat2, nullptr, yTmp, OO);

  instnorm_relu<false><<<BB * OO, 256, 0, stream>>>(yTmp, d_out);
}
